// VectorQuantizer_11579231830280
// MI455X (gfx1250) — compile-verified, matrix operand order checked by reading, not run on hardware
//
#include <hip/hip_runtime.h>
#include <stdint.h>

typedef __attribute__((ext_vector_type(16))) _Float16 v16h;
typedef __attribute__((ext_vector_type(8)))  float    v8f;
typedef __attribute__((ext_vector_type(4)))  unsigned v4u;
typedef __attribute__((ext_vector_type(8)))  int      v8i;
typedef __attribute__((ext_vector_type(4)))  int      v4i;

#define KCODES 1024
#define DDIM   64
#define NROWS  65536
#define ROWS_PER_BLOCK 128
#define CODES_PER_CHUNK 128
#define NCHUNKS (KCODES / CODES_PER_CHUNK)
#define NTILES  (CODES_PER_CHUNK / 16)
#define PADH 72   // halfs per LDS row: 64 data + 8 pad -> 36-dword stride, 16B-aligned frags
#define PADW 36   // dwords per LDS row

#define O_LOSS 0
#define O_ZQ   1
#define O_PERP 4194305
#define O_IDX  4194306
#define NUMEL  4194304

union Frag { v16h v; v4u q[2]; };

// ---------------------------------------------------------------------------
// TDM: DMA one 128x64 f16 codebook chunk Global->LDS with row padding.
// D# packing per CDNA5 ISA ch.8: group0 = {count, lds_addr, global_addr, type=2},
// group1 = {data_size=2B, pad 32dw-interval/4dw-amount, dims 64x128, stride 64}.
// 6-arg toolchain form: (g0, g1, g2, g3, g4, cpol) with unused groups zeroed.
// ---------------------------------------------------------------------------
__device__ __forceinline__ void tdm_load_chunk(const _Float16* gsrc, unsigned ldsAddr)
{
  unsigned long long ga = (unsigned long long)(uintptr_t)gsrc;
  v4u g0;
  g0[0] = 1u;                                               // count=1, no gather
  g0[1] = ldsAddr;                                          // LDS byte address
  g0[2] = (unsigned)(ga & 0xffffffffu);                     // global_addr[31:0]
  g0[3] = (unsigned)((ga >> 32) & 0x1ffffffu) | (2u << 30); // addr[56:32] | type=2
  v8i g1;
  g1[0] = (1 << 16)        // data_size = 2 bytes
        | (1 << 20)        // pad_enable
        | (4 << 22)        // pad_interval: 32 dwords (one 64-half row)
        | (3 << 25);       // pad_amount: 4 dwords (8 halfs -> 72-half stride)
  g1[1] = (int)(64u  << 16);   // tensor_dim0 = 64  (bits 79:48 low part)
  g1[2] = (int)(128u << 16);   // tensor_dim1 = 128 (bits 111:80 low part)
  g1[3] = (int)(64u  << 16);   // tile_dim0 = 64    (bits 127:112)
  g1[4] = 128;                 // tile_dim1 = 128
  g1[5] = 64;                  // tensor_dim0_stride = 64 elements
  g1[6] = 0;
  g1[7] = 0;
  v4i z4 = {0, 0, 0, 0};
  v8i z8 = {0, 0, 0, 0, 0, 0, 0, 0};
  __builtin_amdgcn_tensor_load_to_lds(g0, g1, z4, z4, z8, 0);
}

// ---------------------------------------------------------------------------
// Kernel 0: codebook norms, f16 codebook copy, zero histogram/loss accumulator
// ---------------------------------------------------------------------------
__global__ void __launch_bounds__(1024)
vq_init_kernel(const float* __restrict__ emb, float* __restrict__ eNorm,
               _Float16* __restrict__ embH, int* __restrict__ hist,
               float* __restrict__ lossAcc)
{
  int k = threadIdx.x;
  float s = 0.0f;
#pragma unroll 8
  for (int d = 0; d < DDIM; ++d) {
    float v = emb[k * DDIM + d];
    embH[k * DDIM + d] = (_Float16)v;
    s += v * v;
  }
  eNorm[k] = s;
  hist[k] = 0;
  if (k == 0) *lossAcc = 0.0f;
}

// ---------------------------------------------------------------------------
// Kernel 1: WMMA distance GEMM + per-row argmin + histogram.
//   A tile holds -2*z (f16), C initialized to ||e_k||^2 broadcast, so the
//   WMMA chain directly yields t = ||e||^2 - 2 z.e  (per-row ||z||^2 dropped).
//   B chunks streamed by the Tensor Data Mover, double-buffered (TENSORcnt).
// ---------------------------------------------------------------------------
__global__ void __launch_bounds__(256)
vq_argmin_kernel(const float* __restrict__ z, const _Float16* __restrict__ embH,
                 const float* __restrict__ eNorm, int* __restrict__ idxOut,
                 int* __restrict__ hist)
{
  __shared__ _Float16 Alds[ROWS_PER_BLOCK * PADH];          // 18.0 KB
  __shared__ _Float16 Blds[2][CODES_PER_CHUNK * PADH];      // 36.0 KB (double buf)

  const int tid    = threadIdx.x;
  const int lane   = tid & 31;
  const int wv     = tid >> 5;       // wave 0..7, each owns a 16-row M tile
  const int laneLo = lane & 15;
  const int laneHi = lane >> 4;

  const int nBase = blockIdx.x * ROWS_PER_BLOCK;  // 512 blocks cover N=65536
  const int b     = nBase >> 10;                  // batch (tiles never straddle b)
  const int pBase = nBase & 1023;                 // position within H*W

  // kick off TDM for chunk 0 while every wave builds the A tile
  if (wv == 0)
    tdm_load_chunk(embH, (unsigned)(uintptr_t)&Blds[0][0]);

  // ---- transpose-gather 128 zf rows from [B,D,H,W] into LDS as -2*z (f16) ----
  for (int i = tid; i < ROWS_PER_BLOCK * DDIM; i += 256) {
    int d = i >> 7;          // 0..63
    int r = i & 127;         // row within tile -> coalesced 512B runs
    float v = z[(b << 16) + (d << 10) + pBase + r];
    Alds[r * PADH + d] = (_Float16)(-2.0f * v);
  }
  __syncthreads();

  // ---- gather A fragments once (ISA 7.12.2 16-bit A layout), reuse all K ----
  const unsigned* Au = (const unsigned*)Alds;
  Frag afrag[2];
  {
    int base = (wv * 16 + laneLo) * PADW + laneHi * 4;
#pragma unroll
    for (int h = 0; h < 2; ++h) {
      const v4u* p = (const v4u*)(Au + base + h * 16);   // 16B aligned (PADW%4==0)
      afrag[h].q[0] = p[0];                              // dwords 0..3  (K 0..7)
      afrag[h].q[1] = p[2];                              // dwords 8..11 (K 16..23)
    }
  }

  float bestT[8];
  int   bestI[8];
#pragma unroll
  for (int r = 0; r < 8; ++r) { bestT[r] = 3.0e38f; bestI[r] = 0; }

  for (int chunk = 0; chunk < NCHUNKS; ++chunk) {
    const int buf      = chunk & 1;
    const int codeBase = chunk * CODES_PER_CHUNK;

    // batch the chunk's 8 code norms: loads issue together, no full-drain
    // loadcnt waits inside the tile loop
    float enArr[NTILES];
#pragma unroll
    for (int t = 0; t < NTILES; ++t)
      enArr[t] = eNorm[codeBase + t * 16 + laneLo];

    if (wv == 0) {
      if (chunk + 1 < NCHUNKS) {
        tdm_load_chunk(embH + (chunk + 1) * CODES_PER_CHUNK * DDIM,
                       (unsigned)(uintptr_t)&Blds[buf ^ 1][0]);
        __builtin_amdgcn_s_wait_tensorcnt(1);   // oldest (this chunk) done
      } else {
        __builtin_amdgcn_s_wait_tensorcnt(0);
      }
    }
    __syncthreads();                            // publish TDM data to all waves

    const unsigned* Bu = (const unsigned*)&Blds[buf][0];
#pragma unroll
    for (int tile = 0; tile < NTILES; ++tile) {
      // B fragment: lane = N column (code), same per-lane K packing as A
      Frag bfrag[2];
      int bbase = (tile * 16 + laneLo) * PADW + laneHi * 4;
#pragma unroll
      for (int h = 0; h < 2; ++h) {
        const v4u* p = (const v4u*)(Bu + bbase + h * 16);
        bfrag[h].q[0] = p[0];
        bfrag[h].q[1] = p[2];
      }

      float en = enArr[tile];
      v8f acc;
#pragma unroll
      for (int r = 0; r < 8; ++r) acc[r] = en;   // C = ||e||^2 broadcast per column

      acc = __builtin_amdgcn_wmma_f32_16x16x32_f16(false, afrag[0].v, false, bfrag[0].v,
                                                   (short)0, acc, false, false);
      acc = __builtin_amdgcn_wmma_f32_16x16x32_f16(false, afrag[1].v, false, bfrag[1].v,
                                                   (short)0, acc, false, false);

      int code = codeBase + tile * 16 + laneLo;
#pragma unroll
      for (int r = 0; r < 8; ++r) {
        float t = acc[r];                        // = ||e||^2 - 2 z.e
        if (t < bestT[r]) { bestT[r] = t; bestI[r] = code; }  // codes increase -> first min
      }
    }
    __syncthreads();                             // buffer may be overwritten next iter
  }

  // ---- argmin across the 16 N-lanes of each half (wave32 shuffles) ----
#pragma unroll
  for (int m = 1; m < 16; m <<= 1) {
#pragma unroll
    for (int r = 0; r < 8; ++r) {
      float ot = __shfl_xor(bestT[r], m, 32);
      int   oi = __shfl_xor(bestI[r], m, 32);
      if (ot < bestT[r] || (ot == bestT[r] && oi < bestI[r])) { bestT[r] = ot; bestI[r] = oi; }
    }
  }
  if (laneLo == 0) {
#pragma unroll
    for (int r = 0; r < 8; ++r) {                // C layout: M = r + 8*laneHi
      int row = nBase + wv * 16 + laneHi * 8 + r;
      idxOut[row] = bestI[r];
      atomicAdd(&hist[bestI[r]], 1);
    }
  }
}

// ---------------------------------------------------------------------------
// Kernel 2: z_q gather (== z_q_st numerically) + fused MSE partial + idx out
// ---------------------------------------------------------------------------
__global__ void __launch_bounds__(256)
vq_output_kernel(const float* __restrict__ z, const float* __restrict__ emb,
                 const int* __restrict__ idxWs, float* __restrict__ out,
                 float* __restrict__ lossAcc)
{
  int e   = blockIdx.x * 256 + threadIdx.x;   // element of [B,D,H,W]
  int b   = e >> 16;
  int rem = e & 65535;
  int d   = rem >> 10;
  int p   = rem & 1023;
  int n   = (b << 10) | p;

  int   k   = idxWs[n];
  float zq  = emb[k * DDIM + d];              // codebook lives in L2
  float dif = zq - z[e];                      // z has identical [B,D,H,W] indexing
  out[O_ZQ + e] = zq;
  if (d == 0) out[O_IDX + n] = (float)k;

  float s = dif * dif;
#pragma unroll
  for (int m = 1; m < 32; m <<= 1) s += __shfl_xor(s, m, 32);

  __shared__ float red[8];
  int lane = threadIdx.x & 31, wv = threadIdx.x >> 5;
  if (lane == 0) red[wv] = s;
  __syncthreads();
  if (threadIdx.x == 0) {
    float t = 0.0f;
#pragma unroll
    for (int i = 0; i < 8; ++i) t += red[i];
    atomicAdd(lossAcc, t);
  }
}

// ---------------------------------------------------------------------------
// Kernel 3: perplexity from histogram + final loss scale
// ---------------------------------------------------------------------------
__global__ void __launch_bounds__(1024)
vq_finalize_kernel(const int* __restrict__ hist, const float* __restrict__ lossAcc,
                   float* __restrict__ out)
{
  int t = threadIdx.x;
  float em   = (float)hist[t] * (1.0f / 65536.0f);
  float term = em * logf(em + 1e-10f);
#pragma unroll
  for (int m = 1; m < 32; m <<= 1) term += __shfl_xor(term, m, 32);

  __shared__ float red[32];
  if ((t & 31) == 0) red[t >> 5] = term;
  __syncthreads();
  if (t < 32) {
    float s = red[t];
#pragma unroll
    for (int m = 1; m < 32; m <<= 1) s += __shfl_xor(s, m, 32);
    if (t == 0) {
      out[O_PERP] = expf(-s);
      out[O_LOSS] = 1.25f * (*lossAcc) * (1.0f / (float)NUMEL);
    }
  }
}

// ---------------------------------------------------------------------------
extern "C" void kernel_launch(void* const* d_in, const int* in_sizes, int n_in,
                              void* d_out, int out_size, void* d_ws, size_t ws_size,
                              hipStream_t stream) {
  const float* z   = (const float*)d_in[0];
  const float* emb = (const float*)d_in[1];
  float*       out = (float*)d_out;

  char*     ws      = (char*)d_ws;
  int*      idxWs   = (int*)ws;                    // 65536 * 4 B
  int*      hist    = (int*)(ws + 262144);         // 1024 * 4 B
  float*    eNorm   = (float*)(ws + 266240);       // 1024 * 4 B
  float*    lossAcc = (float*)(ws + 270336);       // 4 B
  _Float16* embH    = (_Float16*)(ws + 270848);    // 65536 * 2 B (f16 codebook)

  vq_init_kernel<<<1, 1024, 0, stream>>>(emb, eNorm, embH, hist, lossAcc);
  vq_argmin_kernel<<<NROWS / ROWS_PER_BLOCK, 256, 0, stream>>>(z, embH, eNorm, idxWs, hist);
  vq_output_kernel<<<NUMEL / 256, 256, 0, stream>>>(z, emb, idxWs, out, lossAcc);
  vq_finalize_kernel<<<1, 1024, 0, stream>>>(hist, lossAcc, out);
}
